// CrystalGCN_47218870452991
// MI455X (gfx1250) — compile-verified, matrix operand order checked by reading, not run on hardware
//
#include <hip/hip_runtime.h>

// ---------------- constants (CrystalGCN problem sizes) ----------------
constexpr int   cN  = 12000;
constexpr int   cE  = 120000;
constexpr int   cF  = 768;
constexpr int   cFE = 64;
constexpr int   cL  = 16;

// ---------------- WMMA fragment types ----------------
typedef __attribute__((ext_vector_type(16))) __bf16 v16bf;
typedef __attribute__((ext_vector_type(8)))  float  v8f;

union FragU { v16bf v; uint4 q[2]; };

__device__ __forceinline__ unsigned short f2bf(float x) {
  union { float f; unsigned int u; } v; v.f = x;
  unsigned int r = v.u + 0x7FFFu + ((v.u >> 16) & 1u);   // round-to-nearest-even
  return (unsigned short)(r >> 16);
}
__device__ __forceinline__ float bf2f(unsigned short h) {
  union { unsigned int u; float f; } v; v.u = ((unsigned int)h) << 16;
  return v.f;
}
__device__ __forceinline__ float sigmoidf_(float x) { return 1.0f / (1.0f + __expf(-x)); }

// A fragment: 16x32 bf16 tile from row-major [rows, lda] bf16 matrix.
// Lane L<16 : row L,   K {0..7, 16..23};  Lane L>=16: row L-16, K {8..15, 24..31}.
__device__ __forceinline__ v16bf load_a_frag(const unsigned short* A, long long row0,
                                             int k0, int lda, int lane) {
  int r  = lane & 15;
  int kb = (lane >> 4) << 3;
  const unsigned short* p = A + (row0 + r) * (long long)lda + (k0 + kb);
  FragU f;
  f.q[0] = *reinterpret_cast<const uint4*>(p);        // K kb..kb+7
  f.q[1] = *reinterpret_cast<const uint4*>(p + 16);   // K kb+16..kb+23
  return f.v;
}

// B fragment from pre-packed weights: block (ntG,kt) is 512 bf16 = 1 KB,
// lane-major, 16 halves per lane (mirror of A layout with N=lane&15).
__device__ __forceinline__ v16bf load_b_frag(const unsigned short* Bp, int ntG,
                                             int kt, int KT, int lane) {
  const unsigned short* p = Bp + (((size_t)ntG * KT + kt) << 9) + ((size_t)lane << 4);
  FragU f;
  f.q[0] = *reinterpret_cast<const uint4*>(p);
  f.q[1] = *reinterpret_cast<const uint4*>(p + 8);
  return f.v;
}

__device__ __forceinline__ v8f wmma_bf16(v16bf a, v16bf b, v8f c) {
  return __builtin_amdgcn_wmma_f32_16x16x32_bf16(false, a, false, b, (short)0, c, false, false);
}

// ---------------- weight packing: fp32 [K,768] slice -> WMMA B layout ----------------
__global__ void pack_w_kernel(const float* __restrict__ W, int ldw, int K,
                              unsigned short* __restrict__ dst, int coff16) {
  int total = K * cF;
  int KT = K >> 5;
  for (int idx = blockIdx.x * blockDim.x + threadIdx.x; idx < total;
       idx += gridDim.x * blockDim.x) {
    int h   = idx & 15;
    int l   = (idx >> 4) & 31;
    int blk = idx >> 9;
    int kt  = blk % KT;
    int ntl = blk / KT;
    int nn  = l & 15;
    int kb  = (l >> 4) << 3;
    int kk  = (h < 8) ? (kb + h) : (kb + 8 + h);       // h>=8 -> kb+16+(h-8)
    int kIdx = (kt << 5) + kk;
    int nIdx = (ntl << 4) + nn;
    size_t db = ((size_t)(ntl + coff16) * KT + kt) << 9;
    dst[db + ((size_t)l << 4) + h] = f2bf(W[(size_t)kIdx * ldw + nIdx]);
  }
}

// ---------------- elementwise helpers ----------------
__global__ void cvt_bf16_kernel(const float* __restrict__ in,
                                unsigned short* __restrict__ out, long long n) {
  long long i = (long long)blockIdx.x * blockDim.x + threadIdx.x;
  long long s = (long long)gridDim.x * blockDim.x;
  for (; i < n; i += s) out[i] = f2bf(in[i]);
}
__global__ void copy_f32_kernel(const float* __restrict__ in,
                                float* __restrict__ out, long long n) {
  long long i = (long long)blockIdx.x * blockDim.x + threadIdx.x;
  long long s = (long long)gridDim.x * blockDim.x;
  for (; i < n; i += s) out[i] = in[i];
}

// ---------------- generic bf16 GEMM, fp32 output (node partials) ----------------
// block = 256 thr = 8 waves; wave tile 16x64; block tile 64 rows x 128 cols.
__global__ __launch_bounds__(256)
void gemm_bf16_f32_kernel(const unsigned short* __restrict__ A, int lda, int M,
                          const unsigned short* __restrict__ Bp, int KT,
                          float* __restrict__ C, int ldc) {
  int lane = threadIdx.x & 31;
  int wave = threadIdx.x >> 5;
  int rg = wave >> 1, cg = wave & 1;
  long long row0 = (long long)blockIdx.y * 64 + rg * 16;
  int col0 = blockIdx.x * 128 + cg * 64;
  if (row0 >= M) return;                 // wave-uniform; 16 | M so no partial tiles
  v8f a0 = {0,0,0,0,0,0,0,0}, a1 = a0, a2 = a0, a3 = a0;
  int nt0 = col0 >> 4;
  for (int kt = 0; kt < KT; ++kt) {
    v16bf a = load_a_frag(A, row0, kt << 5, lda, lane);
    a0 = wmma_bf16(a, load_b_frag(Bp, nt0 + 0, kt, KT, lane), a0);
    a1 = wmma_bf16(a, load_b_frag(Bp, nt0 + 1, kt, KT, lane), a1);
    a2 = wmma_bf16(a, load_b_frag(Bp, nt0 + 2, kt, KT, lane), a2);
    a3 = wmma_bf16(a, load_b_frag(Bp, nt0 + 3, kt, KT, lane), a3);
  }
  int hi = lane >> 4, c = lane & 15;
  v8f accs[4] = {a0, a1, a2, a3};
  for (int j = 0; j < 4; ++j) {
    int col = col0 + j * 16 + c;
#pragma unroll
    for (int vi = 0; vi < 8; ++vi) {
      long long r = row0 + vi + hi * 8;
      C[r * (long long)ldc + col] = accs[j][vi];
    }
  }
}

// ---------------- pre_edge GEMM: tanh(e_raw@W_pre + b) -> bf16 ----------------
__global__ __launch_bounds__(256)
void gemm_tanh_bf16_kernel(const unsigned short* __restrict__ A, int lda, int M,
                           const unsigned short* __restrict__ Bp, int KT,
                           const float* __restrict__ bias,
                           unsigned short* __restrict__ Out /* [M,768] bf16 */) {
  int lane = threadIdx.x & 31;
  int wave = threadIdx.x >> 5;
  int rg = wave >> 1, cg = wave & 1;
  long long row0 = (long long)blockIdx.y * 64 + rg * 16;
  int col0 = blockIdx.x * 128 + cg * 64;
  if (row0 >= M) return;
  v8f a0 = {0,0,0,0,0,0,0,0}, a1 = a0, a2 = a0, a3 = a0;
  int nt0 = col0 >> 4;
  for (int kt = 0; kt < KT; ++kt) {
    v16bf a = load_a_frag(A, row0, kt << 5, lda, lane);
    a0 = wmma_bf16(a, load_b_frag(Bp, nt0 + 0, kt, KT, lane), a0);
    a1 = wmma_bf16(a, load_b_frag(Bp, nt0 + 1, kt, KT, lane), a1);
    a2 = wmma_bf16(a, load_b_frag(Bp, nt0 + 2, kt, KT, lane), a2);
    a3 = wmma_bf16(a, load_b_frag(Bp, nt0 + 3, kt, KT, lane), a3);
  }
  int hi = lane >> 4, c = lane & 15;
  v8f accs[4] = {a0, a1, a2, a3};
  for (int j = 0; j < 4; ++j) {
    int col = col0 + j * 16 + c;
    float bv = bias[col];
#pragma unroll
    for (int vi = 0; vi < 8; ++vi) {
      long long r = row0 + vi + hi * 8;
      Out[r * (long long)cF + col] = f2bf(tanhf(accs[j][vi] + bv));
    }
  }
}

// ---------------- fused edge kernel, layer 1 ----------------
// Computes e@[Wf_e|Ws_e|We_e] tile via WMMA, adds gathered node partials + bias,
// applies relu*sigmoid, scatter-adds message (atomic), writes e2 = e*(1+gate).
__global__ __launch_bounds__(256)
void edge_conv1_kernel(const unsigned short* __restrict__ Ebf,   // e1 bf16 [E,768]
                       const unsigned short* __restrict__ Bp,    // packed [768,2304]
                       const float* __restrict__ P,              // [N,4608] node partials
                       const int* __restrict__ srcIdx, const int* __restrict__ dstIdx,
                       const float* __restrict__ bf, const float* __restrict__ bs,
                       const float* __restrict__ be,
                       float* __restrict__ xOut,                 // [N,768] (pre-init to x)
                       unsigned short* __restrict__ Eout)        // e2 bf16 [E,768]
{
  int lane = threadIdx.x & 31;
  int wave = threadIdx.x >> 5;
  int es = wave >> 2, fs = wave & 3;
  long long edge0 = (long long)blockIdx.x * 32 + es * 16;
  int feat0 = blockIdx.y * 64 + fs * 16;
  int ntf = feat0 >> 4;
  const int KT = cF / 32;
  v8f aF = {0,0,0,0,0,0,0,0}, aS = aF, aE = aF;
  for (int kt = 0; kt < KT; ++kt) {
    v16bf a = load_a_frag(Ebf, edge0, kt << 5, cF, lane);
    aF = wmma_bf16(a, load_b_frag(Bp, ntf,      kt, KT, lane), aF);
    aS = wmma_bf16(a, load_b_frag(Bp, ntf + 48, kt, KT, lane), aS);
    aE = wmma_bf16(a, load_b_frag(Bp, ntf + 96, kt, KT, lane), aE);
  }
  int hi = lane >> 4, c = lane & 15;
  int featG = feat0 + c;
  float bfv = bf[featG], bsv = bs[featG], bev = be[featG];
#pragma unroll
  for (int vi = 0; vi < 8; ++vi) {
    long long edge = edge0 + vi + hi * 8;
    int d = dstIdx[edge], s = srcIdx[edge];
    const float* Pd = P + (size_t)d * 4608;
    const float* Ps = P + (size_t)s * 4608;
    float fsum = aF[vi] + Pd[featG]        + Ps[768 + featG]  + bfv;
    float ssum = aS[vi] + Pd[1536 + featG] + Ps[2304 + featG] + bsv;
    float esum = aE[vi] + Pd[3072 + featG] + Ps[3840 + featG] + bev;
    float msg  = fmaxf(ssum, 0.0f) * sigmoidf_(fsum);
    float gate = sigmoidf_(esum);
    atomicAdd(xOut + (size_t)d * cF + featG, msg);
    size_t ei = (size_t)edge * cF + featG;
    Eout[ei] = f2bf(bf2f(Ebf[ei]) * (1.0f + gate));
  }
}

// ---------------- fused edge kernel, layer 2 (gate path is dead code) ----------------
__global__ __launch_bounds__(256)
void edge_conv2_kernel(const unsigned short* __restrict__ Ebf,   // e2 bf16 [E,768]
                       const unsigned short* __restrict__ Bp,    // packed [768,1536]
                       const float* __restrict__ P,              // [N,3072]
                       const int* __restrict__ srcIdx, const int* __restrict__ dstIdx,
                       const float* __restrict__ bf, const float* __restrict__ bs,
                       float* __restrict__ xOut)                 // [N,768] (pre-init to x1)
{
  int lane = threadIdx.x & 31;
  int wave = threadIdx.x >> 5;
  int es = wave >> 2, fs = wave & 3;
  long long edge0 = (long long)blockIdx.x * 32 + es * 16;
  int feat0 = blockIdx.y * 64 + fs * 16;
  int ntf = feat0 >> 4;
  const int KT = cF / 32;
  v8f aF = {0,0,0,0,0,0,0,0}, aS = aF;
  for (int kt = 0; kt < KT; ++kt) {
    v16bf a = load_a_frag(Ebf, edge0, kt << 5, cF, lane);
    aF = wmma_bf16(a, load_b_frag(Bp, ntf,      kt, KT, lane), aF);
    aS = wmma_bf16(a, load_b_frag(Bp, ntf + 48, kt, KT, lane), aS);
  }
  int hi = lane >> 4, c = lane & 15;
  int featG = feat0 + c;
  float bfv = bf[featG], bsv = bs[featG];
#pragma unroll
  for (int vi = 0; vi < 8; ++vi) {
    long long edge = edge0 + vi + hi * 8;
    int d = dstIdx[edge], s = srcIdx[edge];
    const float* Pd = P + (size_t)d * 3072;
    const float* Ps = P + (size_t)s * 3072;
    float fsum = aF[vi] + Pd[featG]        + Ps[768 + featG]  + bfv;
    float ssum = aS[vi] + Pd[1536 + featG] + Ps[2304 + featG] + bsv;
    float msg  = fmaxf(ssum, 0.0f) * sigmoidf_(fsum);
    atomicAdd(xOut + (size_t)d * cF + featG, msg);
  }
}

// ---------------- pooling + head ----------------
__global__ void colsum_kernel(const float* __restrict__ X, float* __restrict__ pooled) {
  int f = blockIdx.x * blockDim.x + threadIdx.x;
  if (f >= cF) return;
  float s = 0.0f;
  for (int i = 0; i < cN; ++i) s += X[(size_t)i * cF + f];
  pooled[f] = s;
}

__global__ void head_kernel(const float* __restrict__ pooled, const float* __restrict__ Wd,
                            const float* __restrict__ bd, float* __restrict__ out) {
  __shared__ float logits[cL];
  int t = threadIdx.x;
  if (t < cL) {
    float a = bd[t];
    for (int f = 0; f < cF; ++f) a += pooled[f] * Wd[f * cL + t];
    logits[t] = a;
  }
  __syncthreads();
  if (t == 0) {
    float m = logits[0];
    for (int i = 1; i < cL; ++i) m = fmaxf(m, logits[i]);
    float ssum = 0.0f; float ex[cL];
    for (int i = 0; i < cL; ++i) { ex[i] = __expf(logits[i] - m); ssum += ex[i]; }
    for (int i = 0; i < cL; ++i) out[i] = ex[i] / ssum;
  }
}

// ---------------- host-side orchestration ----------------
extern "C" void kernel_launch(void* const* d_in, const int* in_sizes, int n_in,
                              void* d_out, int out_size, void* d_ws, size_t ws_size,
                              hipStream_t stream) {
  (void)in_sizes; (void)n_in; (void)out_size; (void)ws_size;
  const float* x     = (const float*)d_in[0];
  const float* e_raw = (const float*)d_in[1];
  const int*   src   = (const int*)d_in[2];
  const int*   dst   = (const int*)d_in[3];
  const float* W_pre = (const float*)d_in[4];
  const float* b_pre = (const float*)d_in[5];
  const float* Wf1 = (const float*)d_in[6];  const float* bf1 = (const float*)d_in[7];
  const float* Ws1 = (const float*)d_in[8];  const float* bs1 = (const float*)d_in[9];
  const float* We1 = (const float*)d_in[10]; const float* be1 = (const float*)d_in[11];
  const float* Wf2 = (const float*)d_in[12]; const float* bf2 = (const float*)d_in[13];
  const float* Ws2 = (const float*)d_in[14]; const float* bs2 = (const float*)d_in[15];
  const float* Wd  = (const float*)d_in[18]; const float* bd  = (const float*)d_in[19];
  float* out = (float*)d_out;

  char* ws = (char*)d_ws;
  size_t off = 0;
  auto alloc = [&](size_t bytes) -> void* {
    off = (off + 255) & ~(size_t)255;
    void* p = ws + off;
    off += bytes;
    return p;
  };
  unsigned short* e1bf   = (unsigned short*)alloc((size_t)cE * cF * 2);
  unsigned short* e2bf   = (unsigned short*)alloc((size_t)cE * cF * 2);
  unsigned short* xbf    = (unsigned short*)alloc((size_t)cN * cF * 2);
  unsigned short* erawbf = (unsigned short*)alloc((size_t)cE * cFE * 2);
  float* x1   = (float*)alloc((size_t)cN * cF * 4);
  float* x2   = (float*)alloc((size_t)cN * cF * 4);
  float* P    = (float*)alloc((size_t)cN * 4608 * 4);
  unsigned short* NB1 = (unsigned short*)alloc((size_t)cF * 4608 * 2);
  unsigned short* EB1 = (unsigned short*)alloc((size_t)cF * 2304 * 2);
  unsigned short* NB2 = (unsigned short*)alloc((size_t)cF * 3072 * 2);
  unsigned short* EB2 = (unsigned short*)alloc((size_t)cF * 1536 * 2);
  unsigned short* PB  = (unsigned short*)alloc((size_t)cFE * cF * 2);
  float* pooled = (float*)alloc((size_t)cF * 4);

  const int PK = cF * cF;  // 768x768 slice size
  dim3 pkG(1152), pkB(256);
  // pre_edge weight
  pack_w_kernel<<<dim3(192), pkB, 0, stream>>>(W_pre, cF, cFE, PB, 0);
  // layer 1 node-side slices: [f_d | f_s | s_d | s_s | e_d | e_s]
  pack_w_kernel<<<pkG, pkB, 0, stream>>>(Wf1,            cF, cF, NB1, 0);
  pack_w_kernel<<<pkG, pkB, 0, stream>>>(Wf1 + PK,       cF, cF, NB1, 48);
  pack_w_kernel<<<pkG, pkB, 0, stream>>>(Ws1,            cF, cF, NB1, 96);
  pack_w_kernel<<<pkG, pkB, 0, stream>>>(Ws1 + PK,       cF, cF, NB1, 144);
  pack_w_kernel<<<pkG, pkB, 0, stream>>>(We1,            cF, cF, NB1, 192);
  pack_w_kernel<<<pkG, pkB, 0, stream>>>(We1 + PK,       cF, cF, NB1, 240);
  // layer 1 edge-side slices: [f_e | s_e | e_e]
  pack_w_kernel<<<pkG, pkB, 0, stream>>>(Wf1 + 2 * PK,   cF, cF, EB1, 0);
  pack_w_kernel<<<pkG, pkB, 0, stream>>>(Ws1 + 2 * PK,   cF, cF, EB1, 48);
  pack_w_kernel<<<pkG, pkB, 0, stream>>>(We1 + 2 * PK,   cF, cF, EB1, 96);
  // layer 2 node-side: [f_d | f_s | s_d | s_s]   (We2 gate is dead code)
  pack_w_kernel<<<pkG, pkB, 0, stream>>>(Wf2,            cF, cF, NB2, 0);
  pack_w_kernel<<<pkG, pkB, 0, stream>>>(Wf2 + PK,       cF, cF, NB2, 48);
  pack_w_kernel<<<pkG, pkB, 0, stream>>>(Ws2,            cF, cF, NB2, 96);
  pack_w_kernel<<<pkG, pkB, 0, stream>>>(Ws2 + PK,       cF, cF, NB2, 144);
  // layer 2 edge-side: [f_e | s_e]
  pack_w_kernel<<<pkG, pkB, 0, stream>>>(Wf2 + 2 * PK,   cF, cF, EB2, 0);
  pack_w_kernel<<<pkG, pkB, 0, stream>>>(Ws2 + 2 * PK,   cF, cF, EB2, 48);

  // bf16 copies of activations
  cvt_bf16_kernel<<<dim3(2048), dim3(256), 0, stream>>>(e_raw, erawbf, (long long)cE * cFE);
  cvt_bf16_kernel<<<dim3(2048), dim3(256), 0, stream>>>(x, xbf, (long long)cN * cF);

  // e = tanh(e_raw @ W_pre + b_pre)   [E,64]x[64,768]
  gemm_tanh_bf16_kernel<<<dim3(cF / 128, cE / 64), dim3(256), 0, stream>>>(
      erawbf, cFE, cE, PB, cFE / 32, b_pre, e1bf);

  // layer 1 node partials: x @ [Wf_d|Wf_s|Ws_d|Ws_s|We_d|We_s]  -> P [N,4608]
  gemm_bf16_f32_kernel<<<dim3(4608 / 128, (cN + 63) / 64), dim3(256), 0, stream>>>(
      xbf, cF, cN, NB1, cF / 32, P, 4608);

  // x1 = x (agg accumulates into it)
  copy_f32_kernel<<<dim3(2048), dim3(256), 0, stream>>>(x, x1, (long long)cN * cF);

  // fused edge layer 1: WMMA e@W_e + gathers + activations + scatter + e2
  edge_conv1_kernel<<<dim3(cE / 32, cF / 64), dim3(256), 0, stream>>>(
      e1bf, EB1, P, src, dst, bf1, bs1, be1, x1, e2bf);

  // layer 2: node partials of x1
  cvt_bf16_kernel<<<dim3(2048), dim3(256), 0, stream>>>(x1, xbf, (long long)cN * cF);
  gemm_bf16_f32_kernel<<<dim3(3072 / 128, (cN + 63) / 64), dim3(256), 0, stream>>>(
      xbf, cF, cN, NB2, cF / 32, P, 3072);

  copy_f32_kernel<<<dim3(2048), dim3(256), 0, stream>>>(x1, x2, (long long)cN * cF);

  edge_conv2_kernel<<<dim3(cE / 32, cF / 64), dim3(256), 0, stream>>>(
      e2bf, EB2, P, src, dst, bf2, bs2, x2);

  // global sum pool + dense(16) + softmax
  colsum_kernel<<<dim3(3), dim3(256), 0, stream>>>(x2, pooled);
  head_kernel<<<dim3(1), dim3(32), 0, stream>>>(pooled, Wd, bd, out);
}
